// DIN_85925115724376
// MI455X (gfx1250) — matrix operand order repaired and checked by reading, not compiled, not run on hardware
//
#include <hip/hip_runtime.h>
#include <math.h>

// ---------------------------------------------------------------------------
// Types for CDNA5 WMMA (gfx1250, wave32)
// ---------------------------------------------------------------------------
typedef __attribute__((ext_vector_type(16))) __bf16    v16bf;
typedef __attribute__((ext_vector_type(8)))  float     v8f;
typedef __attribute__((ext_vector_type(4)))  unsigned  uvec4;
typedef __attribute__((ext_vector_type(4)))  float     fvec4;

struct FragBits { uvec4 lo, hi; };

// 16 bf16 per lane = two contiguous 16B chunks (ds_load_b128 / global_load_b128)
static __device__ __forceinline__ v16bf load_frag(const __bf16* p) {
    FragBits f;
    f.lo = *(const uvec4*)(p);
    f.hi = *(const uvec4*)(p + 8);
    return __builtin_bit_cast(v16bf, f);
}

static __device__ __forceinline__ v8f wmma_bf16(v16bf a, v16bf b, v8f c) {
    return __builtin_amdgcn_wmma_f32_16x16x32_bf16(false, a, false, b, (short)0, c, false, false);
}

// CDNA5 async global->LDS copy (16B per lane), tracked by ASYNCcnt.
// Flat LDS pointers map to LDS offset in addr[31:0] (ISA 10.2 aperture rules).
static __device__ __forceinline__ void async_ld16(void* lds_dst, const void* gsrc) {
    unsigned l = (unsigned)(uintptr_t)lds_dst;
    asm volatile("global_load_async_to_lds_b128 %0, %1, off"
                 :: "v"(l), "v"(gsrc) : "memory");
}
static __device__ __forceinline__ void wait_async0() {
    asm volatile("s_wait_asynccnt 0x0" ::: "memory");
}

#define Bsz   4096
#define Lh    200
#define LP    208          // 13 M-tiles of 16
#define KA    128          // folded K for attention layer0

// workspace byte offsets
#define OFF_W0T   0u              // 64x128 bf16  (layer0 weights, [n][k])
#define OFF_WC    16384u          // 64x64  f32   (cvec weights,   [k][n])
#define OFF_W1T   32768u          // 32x64  bf16  (layer1 weights, [n][k])
#define OFF_MW0T  36864u          // 256x256 bf16 (main layer0,    [n][k], k padded 244->256)
#define OFF_MW1T  167936u         // 128x256 bf16 (main layer1,    [n][k])
#define OFF_X     233472u         // 4096x256 bf16 (main MLP input rows)

// ---------------------------------------------------------------------------
// Kernel 0: weight preprocessing (fold + transpose + bf16 + K-pad)
// ---------------------------------------------------------------------------
__global__ __launch_bounds__(256) void din_prep(
    const float* __restrict__ aW0, const float* __restrict__ aW1,
    const float* __restrict__ mW0, const float* __restrict__ mW1,
    unsigned short* __restrict__ w0t_u, float* __restrict__ wc,
    unsigned short* __restrict__ w1t_u, unsigned short* __restrict__ mw0t_u,
    unsigned short* __restrict__ mw1t_u)
{
    __bf16* w0t  = (__bf16*)w0t_u;
    __bf16* w1t  = (__bf16*)w1t_u;
    __bf16* mw0t = (__bf16*)mw0t_u;
    __bf16* mw1t = (__bf16*)mw1t_u;
    int g = blockIdx.x * 256 + threadIdx.x;

    if (g < 64 * 128) {         // W0t[n][k]: k<64 -> aW0[k]+aW0[192+k]; else aW0[64+k]
        int n = g >> 7, k = g & 127;
        float v = (k < 64) ? (aW0[k * 64 + n] + aW0[(192 + k) * 64 + n])
                           : aW0[(64 + k) * 64 + n];
        w0t[g] = (__bf16)v;
    }
    if (g < 64 * 64) {          // Wc[k][n] = aW0[64+k][n] - aW0[192+k][n]  (f32)
        int k = g >> 6, n = g & 63;
        wc[g] = aW0[(64 + k) * 64 + n] - aW0[(192 + k) * 64 + n];
    }
    if (g < 32 * 64) {          // W1t[n][k] = aW1[k][n]
        int n = g >> 6, k = g & 63;
        w1t[g] = (__bf16)aW1[k * 32 + n];
    }
    if (g < 256 * 256) {        // mW0t[n][k], K padded 244->256 with zeros
        int n = g >> 8, k = g & 255;
        mw0t[g] = (__bf16)((k < 244) ? mW0[k * 256 + n] : 0.0f);
    }
    if (g < 128 * 256) {        // mW1t[n][k]
        int n = g >> 8, k = g & 255;
        mw1t[g] = (__bf16)mW1[k * 128 + n];
    }
}

// ---------------------------------------------------------------------------
// Kernel 1: per-batch-row attention (one workgroup = one batch element)
//   LDS: A(208x128 bf16) | W0t(64x128) | W1t(32x64) | H1(208x64) + smalls ~103KB
// ---------------------------------------------------------------------------
__global__ __launch_bounds__(256) void din_attn(
    const int* __restrict__ user_id, const int* __restrict__ item_id,
    const int* __restrict__ item_cat, const int* __restrict__ item_dur,
    const float* __restrict__ user_dense, const float* __restrict__ item_dense,
    const int* __restrict__ hist_seq,
    const float* __restrict__ user_W, const float* __restrict__ item_W,
    const float* __restrict__ cat_W, const float* __restrict__ dur_W,
    const float* __restrict__ hist_W,
    const float* __restrict__ ab0, const float* __restrict__ ab1,
    const float* __restrict__ aW2, const float* __restrict__ ab2,
    const unsigned short* __restrict__ w0t_u, const float* __restrict__ wc,
    const unsigned short* __restrict__ w1t_u, unsigned short* __restrict__ X_u)
{
    __shared__ __bf16 shA[LP * KA];        // [hist | hist*t], bf16
    __shared__ __bf16 shW0[64 * KA];       // layer0 weights [n][k]
    __shared__ __bf16 shW1[32 * 64];       // layer1 weights [n][k]
    __shared__ __bf16 shH1[LP * 64];       // relu(layer0) bf16
    __shared__ float  sh_t[64];
    __shared__ float  sh_cvec[64];
    __shared__ int    sh_seq[LP];
    __shared__ float  sh_part[LP][2];      // per-nt partial scores
    __shared__ float  sh_w[LP];            // softmax weights
    __shared__ float  sh_pool4[4][64];
    __shared__ float  sh_pool[64];

    const __bf16* w0t = (const __bf16*)w0t_u;
    const __bf16* w1t = (const __bf16*)w1t_u;
    __bf16*       Xg  = (__bf16*)X_u;

    const int b    = blockIdx.x;
    const int tid  = threadIdx.x;
    const int wave = tid >> 5;
    const int lane = tid & 31;

    // ---- phase A: target embedding, seq ids (+L2 prefetch of gather rows),
    //               async-stage weights into LDS ----------------------------
    if (tid < 64) sh_t[tid] = item_W[(long)item_id[b] * 64 + tid];
    if (tid < LP) {
        int s = (tid < Lh) ? hist_seq[(long)b * Lh + tid] : 0;
        sh_seq[tid] = s;
        if (tid < Lh) __builtin_prefetch(hist_W + (long)s * 64, 0, 3);  // global_prefetch
    }
    for (int i = tid; i < (64 * KA) / 8; i += 256)                // 1024 x 16B
        async_ld16(shW0 + i * 8, w0t + i * 8);
    if (tid < (32 * 64) / 8)                                      // 256 x 16B
        async_ld16(shW1 + tid * 8, w1t + tid * 8);
    wait_async0();
    __syncthreads();

    // ---- phase B: gather hist rows (L2-resident table), build A = [h, h*t] -
    for (int idx = tid; idx < LP * 16; idx += 256) {
        int l = idx >> 4, c = idx & 15;          // c: 4-float chunk within row
        __bf16* dst = &shA[l * KA + c * 4];
        if (l < Lh) {
            long s = sh_seq[l];
            fvec4 h = *(const fvec4*)(hist_W + s * 64 + c * 4);
            float t0 = sh_t[c * 4 + 0], t1 = sh_t[c * 4 + 1];
            float t2 = sh_t[c * 4 + 2], t3 = sh_t[c * 4 + 3];
            dst[0] = (__bf16)h.x;  dst[1] = (__bf16)h.y;
            dst[2] = (__bf16)h.z;  dst[3] = (__bf16)h.w;
            dst[64 + 0] = (__bf16)(h.x * t0);  dst[64 + 1] = (__bf16)(h.y * t1);
            dst[64 + 2] = (__bf16)(h.z * t2);  dst[64 + 3] = (__bf16)(h.w * t3);
        } else {
            for (int q = 0; q < 4; ++q) { dst[q] = (__bf16)0.f; dst[64 + q] = (__bf16)0.f; }
        }
    }
    // per-batch constant vector: cvec[n] = ab0[n] + sum_k t[k]*Wc[k][n]
    if (tid < 64) {
        float acc = ab0[tid];
        for (int k = 0; k < 64; ++k) acc += sh_t[k] * wc[k * 64 + tid];
        sh_cvec[tid] = acc;
    }
    __syncthreads();

    const int kbase = (lane & 16) ? 16 : 0;      // contiguous K split: k = 16*half + s
    const int col   = lane & 15;
    const int roff  = (lane >> 4) * 8;

    // ---- phase C: layer0 WMMA  (208x128 @ 128x64), relu(+cvec) -> H1 bf16 --
    for (int tile = wave; tile < 13 * 4; tile += 8) {
        int mt = tile >> 2, nt = tile & 3;
        int m = mt * 16 + col, n = nt * 16 + col;
        v8f acc = {};
        for (int kt = 0; kt < 4; ++kt) {
            v16bf a  = load_frag(&shA[m * KA + kt * 32 + kbase]);
            v16bf bm = load_frag(&shW0[n * KA + kt * 32 + kbase]);
            acc = wmma_bf16(a, bm, acc);
        }
        float cv = sh_cvec[n];
        for (int r = 0; r < 8; ++r) {
            float h = acc[r] + cv;
            h = h > 0.f ? h : 0.f;
            shH1[(mt * 16 + roff + r) * 64 + n] = (__bf16)h;
        }
    }
    __syncthreads();

    // ---- phase D: layer1 WMMA (208x64 @ 64x32), relu, fold 32->1 scores ----
    for (int tile = wave; tile < 13 * 2; tile += 8) {
        int mt = tile >> 1, nt = tile & 1;
        int m = mt * 16 + col, n = nt * 16 + col;
        v8f acc = {};
        for (int kt = 0; kt < 2; ++kt) {
            v16bf a  = load_frag(&shH1[m * 64 + kt * 32 + kbase]);
            v16bf bm = load_frag(&shW1[n * 64 + kt * 32 + kbase]);
            acc = wmma_bf16(a, bm, acc);
        }
        float bias = ab1[n], a2 = aW2[n];
        for (int r = 0; r < 8; ++r) {
            float h = acc[r] + bias;
            h = (h > 0.f ? h : 0.f) * a2;
            // deterministic sum over the 16 columns of this tile half
            h += __shfl_xor(h, 1); h += __shfl_xor(h, 2);
            h += __shfl_xor(h, 4); h += __shfl_xor(h, 8);
            if (col == 0) sh_part[mt * 16 + roff + r][nt] = h;
        }
    }
    __syncthreads();

    // ---- phase E: masked softmax over 200 (wave 0) -------------------------
    if (wave == 0) {
        float ab2v = ab2[0];
        float sv[7];
        float mx = -3.4e38f;
        for (int i = 0; i < 7; ++i) {
            int l = lane + 32 * i;
            float s;
            if (l < Lh) {
                s = sh_part[l][0] + sh_part[l][1] + ab2v;
                if (sh_seq[l] == 0) s = -1.0e9f;     // pad mask, like reference
            } else {
                s = -3.0e9f;                         // M-padding rows: exp -> 0
            }
            sv[i] = s;
            mx = s > mx ? s : mx;
        }
        mx = fmaxf(mx, __shfl_xor(mx, 16)); mx = fmaxf(mx, __shfl_xor(mx, 8));
        mx = fmaxf(mx, __shfl_xor(mx, 4));  mx = fmaxf(mx, __shfl_xor(mx, 2));
        mx = fmaxf(mx, __shfl_xor(mx, 1));
        float esum = 0.f;
        for (int i = 0; i < 7; ++i) { sv[i] = expf(sv[i] - mx); esum += sv[i]; }
        esum += __shfl_xor(esum, 16); esum += __shfl_xor(esum, 8);
        esum += __shfl_xor(esum, 4);  esum += __shfl_xor(esum, 2);
        esum += __shfl_xor(esum, 1);
        float inv = 1.f / esum;
        for (int i = 0; i < 7; ++i) {
            int l = lane + 32 * i;
            if (l < LP) sh_w[l] = sv[i] * inv;
        }
    }
    __syncthreads();

    // ---- phase F: pooled[j] = sum_l w[l]*hist[l][j], 4-way split over L ----
    {
        int j = tid & 63, q = tid >> 6;              // 200 = 4 x 50
        float acc = 0.f;
        for (int l = q * 50; l < q * 50 + 50; ++l)
            acc += sh_w[l] * (float)shA[l * KA + j];
        sh_pool4[q][j] = acc;
    }
    __syncthreads();
    if (tid < 64)
        sh_pool[tid] = sh_pool4[0][tid] + sh_pool4[1][tid]
                     + sh_pool4[2][tid] + sh_pool4[3][tid];
    __syncthreads();

    // ---- phase G: assemble x row (244 wide, padded to 256) -> workspace ----
    {
        int j = tid;
        float v = 0.f;
        if      (j < 64)  v = user_W[(long)user_id[b] * 64 + j];
        else if (j < 128) v = sh_t[j - 64];
        else if (j < 144) v = cat_W[(long)item_cat[b] * 16 + (j - 128)];
        else if (j < 152) v = dur_W[(long)item_dur[b] * 8 + (j - 144)];
        else if (j < 177) v = user_dense[(long)b * 25 + (j - 152)];
        else if (j < 180) v = item_dense[(long)b * 3 + (j - 177)];
        else if (j < 244) v = sh_pool[j - 180];
        Xg[(long)b * 256 + j] = (__bf16)v;
    }
}

// ---------------------------------------------------------------------------
// Kernel 2: main MLP 256->256->128->1 + sigmoid, 64 rows per block (4 waves)
// ---------------------------------------------------------------------------
__global__ __launch_bounds__(128) void din_mlp(
    const unsigned short* __restrict__ X_u,
    const unsigned short* __restrict__ mw0t_u, const unsigned short* __restrict__ mw1t_u,
    const float* __restrict__ mb0, const float* __restrict__ mb1,
    const float* __restrict__ mW2, const float* __restrict__ mb2,
    float* __restrict__ out)
{
    __shared__ __bf16 shX[64 * 256];   // input tile, later reused as H2 (64x128)
    __shared__ __bf16 shH[64 * 256];   // relu(layer0)

    const __bf16* Xg   = (const __bf16*)X_u;
    const __bf16* mw0t = (const __bf16*)mw0t_u;
    const __bf16* mw1t = (const __bf16*)mw1t_u;

    const int tid  = threadIdx.x;
    const int wave = tid >> 5;
    const int lane = tid & 31;
    const long b0  = (long)blockIdx.x * 64;

    // async-stage 64x256 bf16 input tile (2048 x 16B)
    {
        const __bf16* src = Xg + b0 * 256;
        for (int i = tid; i < (64 * 256) / 8; i += 128)
            async_ld16(shX + i * 8, src + i * 8);
        wait_async0();
    }
    __syncthreads();

    const int kbase = (lane & 16) ? 16 : 0;
    const int col   = lane & 15;
    const int roff  = (lane >> 4) * 8;
    const int m0    = wave * 16;
    const int mrow  = m0 + col;

    // layer0: 64x256 @ 256x256 (B-fragments stream from L2-resident mw0t)
    for (int nt = 0; nt < 16; ++nt) {
        int n = nt * 16 + col;
        v8f acc = {};
        for (int kt = 0; kt < 8; ++kt) {
            v16bf a  = load_frag(&shX[mrow * 256 + kt * 32 + kbase]);
            v16bf bm = load_frag(&mw0t[(long)n * 256 + kt * 32 + kbase]);
            acc = wmma_bf16(a, bm, acc);
        }
        float bias = mb0[n];
        for (int r = 0; r < 8; ++r) {
            float h = acc[r] + bias;
            h = h > 0.f ? h : 0.f;
            shH[(m0 + roff + r) * 256 + n] = (__bf16)h;
        }
    }
    __syncthreads();

    // layer1: 64x256 @ 256x128 -> H2 (reuse shX)
    __bf16* shH2 = shX;
    for (int nt = 0; nt < 8; ++nt) {
        int n = nt * 16 + col;
        v8f acc = {};
        for (int kt = 0; kt < 8; ++kt) {
            v16bf a  = load_frag(&shH[mrow * 256 + kt * 32 + kbase]);
            v16bf bm = load_frag(&mw1t[(long)n * 256 + kt * 32 + kbase]);
            acc = wmma_bf16(a, bm, acc);
        }
        float bias = mb1[n];
        for (int r = 0; r < 8; ++r) {
            float h = acc[r] + bias;
            h = h > 0.f ? h : 0.f;
            shH2[(m0 + roff + r) * 128 + n] = (__bf16)h;
        }
    }
    __syncthreads();

    // layer2: 128-dot + sigmoid
    if (tid < 64) {
        float acc = mb2[0];
        for (int k = 0; k < 128; ++k)
            acc += (float)shH2[tid * 128 + k] * mW2[k];
        out[b0 + tid] = 1.f / (1.f + expf(-acc));
    }
}

// ---------------------------------------------------------------------------
extern "C" void kernel_launch(void* const* d_in, const int* in_sizes, int n_in,
                              void* d_out, int out_size, void* d_ws, size_t ws_size,
                              hipStream_t stream)
{
    const int*   user_id    = (const int*)d_in[0];
    const int*   item_id    = (const int*)d_in[1];
    const int*   item_cat   = (const int*)d_in[2];
    const int*   item_dur   = (const int*)d_in[3];
    const float* user_dense = (const float*)d_in[4];
    const float* item_dense = (const float*)d_in[5];
    const int*   hist_seq   = (const int*)d_in[6];
    const float* user_W     = (const float*)d_in[7];
    const float* item_W     = (const float*)d_in[8];
    const float* cat_W      = (const float*)d_in[9];
    const float* dur_W      = (const float*)d_in[10];
    const float* hist_W     = (const float*)d_in[11];
    const float* aW0 = (const float*)d_in[12];
    const float* ab0 = (const float*)d_in[13];
    const float* aW1 = (const float*)d_in[14];
    const float* ab1 = (const float*)d_in[15];
    const float* aW2 = (const float*)d_in[16];
    const float* ab2 = (const float*)d_in[17];
    const float* mW0 = (const float*)d_in[18];
    const float* mb0 = (const float*)d_in[19];
    const float* mW1 = (const float*)d_in[20];
    const float* mb1 = (const float*)d_in[21];
    const float* mW2 = (const float*)d_in[22];
    const float* mb2 = (const float*)d_in[23];

    char* ws = (char*)d_ws;
    unsigned short* w0t  = (unsigned short*)(ws + OFF_W0T);
    float*          wc   = (float*)(ws + OFF_WC);
    unsigned short* w1t  = (unsigned short*)(ws + OFF_W1T);
    unsigned short* mw0t = (unsigned short*)(ws + OFF_MW0T);
    unsigned short* mw1t = (unsigned short*)(ws + OFF_MW1T);
    unsigned short* Xb   = (unsigned short*)(ws + OFF_X);

    din_prep<<<256, 256, 0, stream>>>(aW0, aW1, mW0, mW1, w0t, wc, w1t, mw0t, mw1t);

    din_attn<<<Bsz, 256, 0, stream>>>(user_id, item_id, item_cat, item_dur,
                                      user_dense, item_dense, hist_seq,
                                      user_W, item_W, cat_W, dur_W, hist_W,
                                      ab0, ab1, aW2, ab2, w0t, wc, w1t, Xb);

    din_mlp<<<Bsz / 64, 128, 0, stream>>>(Xb, mw0t, mw1t, mb0, mb1, mW2, mb2,
                                          (float*)d_out);
}
